// HGraphAttentionLayer_21620865368431
// MI455X (gfx1250) — matrix operand order repaired and checked by reading, not compiled, not run on hardware
//
#include <hip/hip_runtime.h>
#include <hip/hip_bf16.h>

typedef __attribute__((ext_vector_type(16))) __bf16 v16bf;
typedef __attribute__((ext_vector_type(8)))  float  v8f;
typedef __attribute__((ext_vector_type(4)))  int    i4;

namespace {
constexpr int N = 8192, F = 256, H = 4, O = 128, NRNA = 5000;
constexpr int NT16 = N / 16;   // 512 row tiles
constexpr int JT32 = N / 32;   // 256 k tiles for attention GEMM
constexpr int OT16 = O / 16;   // 8 output col tiles
constexpr int FP   = F + 8;    // padded LDS row: 264 floats -> 16B aligned, bank-shifted
constexpr float SLOPE = 0.2f, EPS = 1e-5f;

union AFrag { unsigned u[8]; i4 q[2]; v16bf v; };

__device__ __forceinline__ unsigned short f2bf(float f) {
  unsigned u = __float_as_uint(f);
  u += 0x7FFFu + ((u >> 16) & 1u);          // round-to-nearest-even
  return (unsigned short)(u >> 16);
}
__device__ __forceinline__ unsigned packbf(float a, float b) {
  return (unsigned)f2bf(a) | ((unsigned)f2bf(b) << 16);
}
__device__ __forceinline__ float lrelu(float x) { return x > 0.f ? x : SLOPE * x; }
// 16-bit A-matrix 16x32 per-lane K start (CDNA5 ISA 7.12.2)
__device__ __forceinline__ int a_k0(int lane, int v) {
  return 2 * v + (v >= 4 ? 8 : 0) + (lane >= 16 ? 8 : 0);
}
// 16-bit B-matrix 32x16 per-lane K start: same K order as A (a_k0)
__device__ __forceinline__ int b_k0(int lane, int v) {
  return 2 * v + (v >= 4 ? 8 : 0) + (lane >= 16 ? 8 : 0);
}
} // namespace

// ---------------- k0: zero accumulators (ws is poisoned before timing) -------
__global__ void __launch_bounds__(256) k0_zero(float* vals, float* Dsum, float* stats) {
  int idx = blockIdx.x * 256 + threadIdx.x;
  if (idx < N * O) vals[idx] = 0.f;
  if (idx < H * N) Dsum[idx] = 0.f;
  if (idx < 8)     stats[idx] = 0.f;
}

// ---------------- k1: per-type projection feats[h,n,:] (bf16 WMMA) -----------
__global__ void __launch_bounds__(256) k1_project(const float* __restrict__ input,
    const float* __restrict__ projRna, const float* __restrict__ projDis,
    float* __restrict__ feats) {
  __shared__ float tile[16][FP];
  const int nt = blockIdx.x, h = blockIdx.y, n0 = nt * 16, t = threadIdx.x;
  for (int idx = t; idx < 16 * F; idx += 256) {
    int m = idx >> 8, f = idx & (F - 1);
    tile[m][f] = input[(size_t)(n0 + m) * F + f];
  }
  __syncthreads();
  const int lane = t & 31, wv = t >> 5, M = lane & 15;
  const int oc = wv * 16 + (lane & 15);
  v8f acc = {0.f, 0.f, 0.f, 0.f, 0.f, 0.f, 0.f, 0.f};
  const bool hasRna = (n0 < NRNA), hasDis = (n0 + 15 >= NRNA);
  for (int pass = 0; pass < 2; ++pass) {
    if (pass == 0 && !hasRna) continue;
    if (pass == 1 && !hasDis) continue;
    const float* P = (pass == 0) ? projRna : projDis;
    // branch-free row masking: multiply instead of predicated loads
    const float rowScale =
        ((pass == 0) ? (n0 + M < NRNA) : (n0 + M >= NRNA)) ? 1.f : 0.f;
    for (int kb = 0; kb < F; kb += 32) {
      AFrag a, b;
#pragma unroll
      for (int v = 0; v < 8; ++v) {
        int ka = kb + a_k0(lane, v);
        float2 aa = *(const float2*)&tile[M][ka];       // ds_load_b64
        a.u[v] = packbf(aa.x * rowScale, aa.y * rowScale);
        int kc = kb + b_k0(lane, v);
        float b0 = P[((size_t)h * F + kc) * O + oc];
        float b1 = P[((size_t)h * F + kc + 1) * O + oc];
        b.u[v] = packbf(b0, b1);
      }
      acc = __builtin_amdgcn_wmma_f32_16x16x32_bf16(false, a.v, false, b.v,
                                                    (short)0, acc, false, false);
    }
  }
#pragma unroll
  for (int r = 0; r < 8; ++r) {
    int n = n0 + r + (lane >= 16 ? 8 : 0);
    feats[((size_t)h * N + n) * O + oc] = acc[r];
  }
}

// ---------------- k2: attention score vectors --------------------------------
__global__ void __launch_bounds__(256) k2_scores(const float* __restrict__ feats,
    const float* __restrict__ scS, const float* __restrict__ scT,
    float* __restrict__ sSrc, float* __restrict__ sTgt) {
  int idx = blockIdx.x * 256 + threadIdx.x;     // < H*N by grid
  int h = idx / N, n = idx % N;
  const float* fr = feats + ((size_t)h * N + n) * O;
  float a = 0.f, b = 0.f;
  for (int o = 0; o < O; ++o) { float f = fr[o]; a += f * scS[h * O + o]; b += f * scT[h * O + o]; }
  sSrc[idx] = a; sTgt[idx] = b;
}

// ---------------- k3: softmax column sums D[h,j] over rows i -----------------
__global__ void __launch_bounds__(256) k3_colsum(const float* __restrict__ mask,
    const float* __restrict__ sSrc, const float* __restrict__ sTgt,
    float* __restrict__ Dsum) {
  __shared__ float ls[H][512];
  const int t = threadIdx.x;
  const int j = blockIdx.x * 256 + t;
  const int r0 = blockIdx.y * 1024;
  float st[H], acc[H];
#pragma unroll
  for (int h = 0; h < H; ++h) { st[h] = sTgt[h * N + j]; acc[h] = 0.f; }
  for (int c = 0; c < 2; ++c) {
    int rb = r0 + c * 512;
    __syncthreads();
    for (int idx = t; idx < H * 512; idx += 256) {
      int h = idx >> 9, r = idx & 511;
      ls[h][r] = sSrc[h * N + rb + r];
    }
    __syncthreads();
    for (int r = 0; r < 512; ++r) {
      float m = mask[(size_t)(rb + r) * N + j];   // coalesced over j
#pragma unroll
      for (int h = 0; h < H; ++h)
        acc[h] += __expf(m + lrelu(ls[h][r] + st[h]));  // masked -> exp(-1e9)=0
    }
  }
#pragma unroll
  for (int h = 0; h < H; ++h) atomicAdd(&Dsum[h * N + j], acc[h]);
}

// ---------------- k4: stage feats/D in WMMA B-fragment layout (bf16) ---------
__global__ void __launch_bounds__(256) k4_stageB(const float* __restrict__ feats,
    const float* __restrict__ Dsum, unsigned* __restrict__ featsB) {
  const int t = threadIdx.x, lane = t & 31;
  const int tileIdx = blockIdx.x * 8 + (t >> 5);       // ((h*JT32+jt)*OT16+ot)
  const int ot = tileIdx & 7, jt = (tileIdx >> 3) & (JT32 - 1), h = tileIdx >> 11;
  const int oc = ot * 16 + (lane & 15);
  unsigned* dst = featsB + (size_t)tileIdx * 256 + lane * 8;
#pragma unroll
  for (int v = 0; v < 8; ++v) {
    int j = jt * 32 + b_k0(lane, v);
    float d0 = 1.0f / Dsum[h * N + j];
    float d1 = 1.0f / Dsum[h * N + j + 1];
    float b0 = feats[((size_t)h * N + j) * O + oc] * d0;
    float b1 = feats[((size_t)h * N + j + 1) * O + oc] * d1;
    dst[v] = packbf(b0, b1);
  }
}

// ---------------- k5: fused exp-numerator + WMMA aggregation GEMM ------------
// Mask stream (the dominant HBM traffic) is double-buffered through LDS with
// gfx1250 async-to-LDS loads so the fetch of j-tile jt+1 overlaps the exp +
// v_wmma work on tile jt.
__global__ void __launch_bounds__(256) k5_attn(const float* __restrict__ mask,
    const float* __restrict__ sSrc, const float* __restrict__ sTgt,
    const unsigned* __restrict__ featsB, float* __restrict__ vals) {
  __shared__ unsigned aTile[32][8];                    // A fragment, shared by 8 waves
  __shared__ float2 mStage[2][256];                    // async mask staging (2x2KB)
  const int t = threadIdx.x, ib = blockIdx.x, h = blockIdx.y;
  // producer slot: each thread owns one packed dword of the A fragment
  const int pLane = t >> 3, pv = t & 7;
  const int pM = pLane & 15;
  const int pk0 = a_k0(pLane, pv);
  const int i = ib * 16 + pM;
  const float ss = sSrc[h * N + i];
  const float* rowBase = mask + (size_t)i * N + pk0;   // this thread's mask stream
  const float* stBase  = sTgt + (size_t)h * N + pk0;
  const unsigned ldsOff0 = (unsigned)(size_t)&mStage[0][t];
  const unsigned ldsOff1 = (unsigned)(size_t)&mStage[1][t];
  // consumer: wave wv handles output column tile wv
  const int lane = t & 31, wv = t >> 5;
  const i4* bPtr = (const i4*)(featsB + ((size_t)h * JT32 * OT16 + wv) * 256 + lane * 8);
  v8f acc = {0.f, 0.f, 0.f, 0.f, 0.f, 0.f, 0.f, 0.f};
  // prime buffer 0 with j-tile 0
  asm volatile("global_load_async_to_lds_b64 %0, %1, off"
               :: "v"(ldsOff0), "v"((unsigned long long)(size_t)rowBase) : "memory");
  for (int jt = 0; jt < JT32; ++jt) {
    // issue next tile into the other buffer (index wraps; harmless refetch at end)
    const unsigned long long gNext =
        (unsigned long long)(size_t)(rowBase + ((jt + 1) & (JT32 - 1)) * 32);
    const unsigned nextLds = (jt & 1) ? ldsOff0 : ldsOff1;
    asm volatile("global_load_async_to_lds_b64 %0, %1, off"
                 :: "v"(nextLds), "v"(gNext) : "memory");
    asm volatile("s_wait_asynccnt 0x1" ::: "memory");  // current buffer landed
    float2 mm = mStage[jt & 1][t];                     // ds_load_b64
    float2 tt = *(const float2*)(stBase + jt * 32);
    float e0 = __expf(mm.x + lrelu(ss + tt.x));
    float e1 = __expf(mm.y + lrelu(ss + tt.y));
    aTile[pLane][pv] = packbf(e0, e1);
    __syncthreads();
    AFrag a, b;
    const i4* ap = (const i4*)&aTile[lane][0];
    a.q[0] = ap[0]; a.q[1] = ap[1];                    // ds_load_b128 x2
    b.q[0] = bPtr[0]; b.q[1] = bPtr[1];                // global_load_b128 x2 (L2-hot)
    bPtr += (OT16 * 256) / 4;                          // advance one j-step
    acc = __builtin_amdgcn_wmma_f32_16x16x32_bf16(false, a.v, false, b.v,
                                                  (short)0, acc, false, false);
    __syncthreads();
  }
#pragma unroll
  for (int r = 0; r < 8; ++r) {
    int n = ib * 16 + r + (lane >= 16 ? 8 : 0);
    int oc = wv * 16 + (lane & 15);
    atomicAdd(&vals[(size_t)n * O + oc], 0.25f * acc[r]);  // head mean
  }
}

// ---------------- k6: global mean / var reduction ----------------------------
__global__ void __launch_bounds__(256) k6_stats(const float* __restrict__ vals,
                                                float* __restrict__ stats) {
  int idx0 = (blockIdx.x * 256 + threadIdx.x) * 8;
  float s = 0.f, s2 = 0.f;
#pragma unroll
  for (int k = 0; k < 8; ++k) { float v = vals[idx0 + k]; s += v; s2 += v * v; }
#pragma unroll
  for (int off = 16; off > 0; off >>= 1) {
    s  += __shfl_down(s, off, 32);
    s2 += __shfl_down(s2, off, 32);
  }
  if ((threadIdx.x & 31) == 0) { atomicAdd(&stats[0], s); atomicAdd(&stats[1], s2); }
}

// ---------------- k7: residual GEMM (WMMA) + instance-norm + ELU -------------
__global__ void __launch_bounds__(256) k7_final(const float* __restrict__ input,
    const float* __restrict__ Rw, const float* __restrict__ vals,
    const float* __restrict__ stats, float* __restrict__ out) {
  __shared__ float tile[16][FP];
  const int nt = blockIdx.x, n0 = nt * 16, t = threadIdx.x;
  for (int idx = t; idx < 16 * F; idx += 256) {
    int m = idx >> 8, f = idx & (F - 1);
    tile[m][f] = input[(size_t)(n0 + m) * F + f];
  }
  __syncthreads();
  const int lane = t & 31, wv = t >> 5, M = lane & 15;
  const int oc = wv * 16 + (lane & 15);
  v8f acc = {0.f, 0.f, 0.f, 0.f, 0.f, 0.f, 0.f, 0.f};
  for (int kb = 0; kb < F; kb += 32) {
    AFrag a, b;
#pragma unroll
    for (int v = 0; v < 8; ++v) {
      int ka = kb + a_k0(lane, v);
      float2 aa = *(const float2*)&tile[M][ka];                // ds_load_b64
      a.u[v] = packbf(aa.x, aa.y);
      int kc = kb + b_k0(lane, v);
      float2 bb = *(const float2*)(Rw + (size_t)oc * F + kc);  // Rw.T gather
      b.u[v] = packbf(bb.x, bb.y);
    }
    acc = __builtin_amdgcn_wmma_f32_16x16x32_bf16(false, a.v, false, b.v,
                                                  (short)0, acc, false, false);
  }
  const float inv = 1.0f / (float)(N * O);
  const float mean = stats[0] * inv;
  const float var = stats[1] * inv - mean * mean;
  const float rstd = rsqrtf(var + EPS);
#pragma unroll
  for (int r = 0; r < 8; ++r) {
    int n = n0 + r + (lane >= 16 ? 8 : 0);
    float x = (vals[(size_t)n * O + oc] - mean) * rstd + acc[r];
    out[(size_t)n * O + oc] = x > 0.f ? x : __expf(x) - 1.f;   // ELU
  }
}

extern "C" void kernel_launch(void* const* d_in, const int* in_sizes, int n_in,
                              void* d_out, int out_size, void* d_ws, size_t ws_size,
                              hipStream_t stream) {
  (void)in_sizes; (void)n_in; (void)out_size; (void)ws_size;
  const float* input = (const float*)d_in[0];
  const float* mask  = (const float*)d_in[1];
  const float* projR = (const float*)d_in[2];
  const float* projD = (const float*)d_in[3];
  const float* scS   = (const float*)d_in[4];
  const float* scT   = (const float*)d_in[5];
  const float* Rw    = (const float*)d_in[6];
  float* out = (float*)d_out;

  // workspace layout (~30 MB)
  float* feats = (float*)d_ws;                       // H*N*O f32
  float* sSrc  = feats + (size_t)H * N * O;          // H*N
  float* sTgt  = sSrc + H * N;                       // H*N
  float* Dsum  = sTgt + H * N;                       // H*N
  float* vals  = Dsum + H * N;                       // N*O
  float* stats = vals + (size_t)N * O;               // 8 (sum, sumsq, pad)
  unsigned* featsB = (unsigned*)(stats + 8);         // H*N*O bf16 (B-layout)

  hipLaunchKernelGGL(k0_zero,   dim3((N * O) / 256), dim3(256), 0, stream, vals, Dsum, stats);
  hipLaunchKernelGGL(k1_project, dim3(NT16, H),      dim3(256), 0, stream, input, projR, projD, feats);
  hipLaunchKernelGGL(k2_scores, dim3((H * N) / 256), dim3(256), 0, stream, feats, scS, scT, sSrc, sTgt);
  hipLaunchKernelGGL(k3_colsum, dim3(N / 256, 8),    dim3(256), 0, stream, mask, sSrc, sTgt, Dsum);
  hipLaunchKernelGGL(k4_stageB, dim3(H * JT32),      dim3(256), 0, stream, feats, Dsum, featsB);
  hipLaunchKernelGGL(k5_attn,   dim3(NT16, H),       dim3(256), 0, stream, mask, sSrc, sTgt, featsB, vals);
  hipLaunchKernelGGL(k6_stats,  dim3(N * O / 2048),  dim3(256), 0, stream, vals, stats);
  hipLaunchKernelGGL(k7_final,  dim3(NT16),          dim3(256), 0, stream, input, Rw, vals, stats, out);
}
